// PCN_38336878084190
// MI455X (gfx1250) — compile-verified, matrix operand order checked by reading, not run on hardware
//
#include <hip/hip_runtime.h>
#include <hip/hip_bf16.h>

// ---------------------------------------------------------------------------
// Fused 3-layer MLP with on-the-fly triangle-wave weights.
//   B = 131072 rows, width 256, param dim 20.
// bf16 WMMA GEMM core; weights streamed into double-buffered LDS chunks by
// the Tensor Data Mover (TENSORcnt-pipelined). Each wave owns 32 rows
// (2 A-strips) so every LDS B-fragment feeds two independent WMMAs.
// ---------------------------------------------------------------------------

typedef __bf16 bf16;
typedef __attribute__((ext_vector_type(16))) __bf16 v16bf;
typedef __attribute__((ext_vector_type(8)))  __bf16 v8bf;
typedef __attribute__((ext_vector_type(8)))  float  v8f;
typedef __attribute__((ext_vector_type(4)))  unsigned int v4u;
typedef __attribute__((ext_vector_type(8)))  int v8i;
typedef __attribute__((ext_vector_type(4)))  int v4i;

#define B_ROWS 131072
#define LW     256          // layer width
#define PD     20           // param dim
#define NT     16           // n tiles  (256 / 16)
#define KS     8            // k steps  (256 / 32)
#define W_ELEMS_PER_LAYER (NT * KS * 32 * 16)    // 65536 bf16 = 128 KB
#define CHUNK_ELEMS       (W_ELEMS_PER_LAYER/2)  // 32768 bf16 = 64 KB (8 tiles)

// ---------------------------------------------------------------------------
// Kernel 1: build W matrices directly in WMMA B-fragment layout.
//   slot e<8  -> K = kstep*32 + h*8 + e
//   slot e>=8 -> K = kstep*32 + 16 + h*8 + (e-8)     (h = lane>>4)
//   column    N = ntile*16 + (lane & 15)
// ---------------------------------------------------------------------------
__global__ void pcn_build_w(const float* __restrict__ p0,
                            const float* __restrict__ p1,
                            const float* __restrict__ p2,
                            const float* __restrict__ p3,
                            bf16* __restrict__ wf) {
    int idx = blockIdx.x * blockDim.x + threadIdx.x;
    const int total = 3 * W_ELEMS_PER_LAYER;
    if (idx >= total) return;

    int r = idx;
    const int e    = r & 15; r >>= 4;
    const int lane = r & 31; r >>= 5;
    const int k    = r & 7;  r >>= 3;
    const int n    = r & 15; r >>= 4;
    const int layer = r;     // 0..2

    const int h   = lane >> 4;
    const int col = n * 16 + (lane & 15);
    const int kk  = k * 32 + ((e < 8) ? (h * 8 + e) : (16 + h * 8 + (e - 8)));

    const float* pa = (layer == 0) ? p0 : (layer == 1) ? p1 : p2;
    const float* pb = (layer == 0) ? p1 : (layer == 1) ? p2 : p3;
    pa += kk  * PD;
    pb += col * PD;

    float d2 = 0.0f;
#pragma unroll
    for (int i = 0; i < PD; ++i) {
        float df = pa[i] - pb[i];
        d2 += df * df;
    }
    float dist = sqrtf(fmaxf(d2, 0.0f));
    float md = fmodf(dist, 0.2f);
    float w  = 10.0f * (0.05f - fabsf(md - 0.1f)) * 0.0625f;   // * 1/sqrt(256)

    wf[idx] = (bf16)w;
}

// ---------------------------------------------------------------------------
// TDM: DMA one 64 KB weight chunk (contiguous) global -> LDS.
// ---------------------------------------------------------------------------
__device__ __forceinline__ void tdm_load_chunk(const bf16* gsrc, bf16* ldst) {
    const unsigned long long ga = (unsigned long long)(uintptr_t)gsrc;
    const unsigned lds = (unsigned)(uintptr_t)ldst;   // low 32 bits = LDS offset

    v4u g0;
    g0.x = 1u;                                         // count=1, user D#
    g0.y = lds;                                        // lds_addr [63:32]
    g0.z = (unsigned)(ga & 0xFFFFFFFFu);               // global_addr[31:0]
    g0.w = (unsigned)((ga >> 32) & 0x01FFFFFFu)        // global_addr[56:32]
         | (2u << 30);                                 // type = 2 ("image")

    v8i g1;
    g1[0] = (int)(3u << 16);          // workgroup_mask=0, data_size=3 (8B)
    g1[1] = (int)(8192u << 16);       // tensor_dim0[15:0]=8192 in [31:16]
    g1[2] = (int)(1u << 16);          // tensor_dim0[31:16]=0; tensor_dim1=1
    g1[3] = (int)(8192u << 16);       // tensor_dim1[31:16]=0; tile_dim0=8192
    g1[4] = 0;                        // tile_dim1=0 (1-D), tile_dim2=0
    g1[5] = 8192;                     // tensor_dim0_stride[31:0]
    g1[6] = 0;
    g1[7] = 0;

    v4i z4 = {0, 0, 0, 0};
    v8i z8 = {0, 0, 0, 0, 0, 0, 0, 0};
    __builtin_amdgcn_tensor_load_to_lds(g0, g1, z4, z4, z8, 0);
}

// ---------------------------------------------------------------------------
// 8 output tiles of one chunk, two 16-row strips per wave:
//   relu(A{0,1} @ Wchunk + b) -> wave LDS slab (strip1 at row offset 16).
// Each B fragment load feeds two independent WMMAs.
// ---------------------------------------------------------------------------
template<int TBASE>
__device__ __forceinline__ void chunk_relu_lds(const bf16* __restrict__ wbuf,
                                               const float* __restrict__ br,
                                               const v16bf* __restrict__ afA,
                                               const v16bf* __restrict__ afB,
                                               bf16* __restrict__ zw, int lane) {
    const int m = lane & 15, h = lane >> 4;
    const v16bf* wl = (const v16bf*)wbuf;
#pragma unroll 1
    for (int t = 0; t < 8; ++t) {
        const float bv = br[TBASE + t];
        v8f acc0 = {bv, bv, bv, bv, bv, bv, bv, bv};
        v8f acc1 = acc0;
#pragma unroll
        for (int k = 0; k < KS; ++k) {
            v16bf bm = wl[(t * KS + k) * 32 + lane];
            acc0 = __builtin_amdgcn_wmma_f32_16x16x32_bf16(
                false, afA[k], false, bm, (short)0, acc0, false, false);
            acc1 = __builtin_amdgcn_wmma_f32_16x16x32_bf16(
                false, afB[k], false, bm, (short)0, acc1, false, false);
        }
#pragma unroll
        for (int j = 0; j < 8; ++j) {
            zw[(j + 8 * h) * LW + (TBASE + t) * 16 + m] =
                (bf16)fmaxf(acc0[j], 0.0f);
            zw[(16 + j + 8 * h) * LW + (TBASE + t) * 16 + m] =
                (bf16)fmaxf(acc1[j], 0.0f);
        }
    }
}

// Final layer: two strips -> f32 global output.
template<int TBASE>
__device__ __forceinline__ void chunk_out(const bf16* __restrict__ wbuf,
                                          const float* __restrict__ br,
                                          const v16bf* __restrict__ afA,
                                          const v16bf* __restrict__ afB,
                                          float* __restrict__ out,
                                          int row0, int lane) {
    const int m = lane & 15, h = lane >> 4;
    const v16bf* wl = (const v16bf*)wbuf;
#pragma unroll 1
    for (int t = 0; t < 8; ++t) {
        const float bv = br[TBASE + t];
        v8f acc0 = {bv, bv, bv, bv, bv, bv, bv, bv};
        v8f acc1 = acc0;
#pragma unroll
        for (int k = 0; k < KS; ++k) {
            v16bf bm = wl[(t * KS + k) * 32 + lane];
            acc0 = __builtin_amdgcn_wmma_f32_16x16x32_bf16(
                false, afA[k], false, bm, (short)0, acc0, false, false);
            acc1 = __builtin_amdgcn_wmma_f32_16x16x32_bf16(
                false, afB[k], false, bm, (short)0, acc1, false, false);
        }
#pragma unroll
        for (int j = 0; j < 8; ++j) {
            out[(size_t)(row0 + j + 8 * h) * LW + (TBASE + t) * 16 + m] = acc0[j];
            out[(size_t)(row0 + 16 + j + 8 * h) * LW + (TBASE + t) * 16 + m] = acc1[j];
        }
    }
}

// Build one strip's A fragments from f32 x rows (global, bf16 convert).
__device__ __forceinline__ void build_afrag_x(v16bf* af, const float* __restrict__ x,
                                              int row, int lane) {
    const int m = lane & 15, h = lane >> 4;
    const float* xr = x + (size_t)(row + m) * LW;
#pragma unroll
    for (int k = 0; k < KS; ++k) {
        union { v16bf v; bf16 e[16]; } a;
        const float4 f0 = *(const float4*)(xr + k * 32 + h * 8);
        const float4 f1 = *(const float4*)(xr + k * 32 + h * 8 + 4);
        const float4 f2 = *(const float4*)(xr + k * 32 + 16 + h * 8);
        const float4 f3 = *(const float4*)(xr + k * 32 + 16 + h * 8 + 4);
        a.e[0]  = (bf16)f0.x; a.e[1]  = (bf16)f0.y;
        a.e[2]  = (bf16)f0.z; a.e[3]  = (bf16)f0.w;
        a.e[4]  = (bf16)f1.x; a.e[5]  = (bf16)f1.y;
        a.e[6]  = (bf16)f1.z; a.e[7]  = (bf16)f1.w;
        a.e[8]  = (bf16)f2.x; a.e[9]  = (bf16)f2.y;
        a.e[10] = (bf16)f2.z; a.e[11] = (bf16)f2.w;
        a.e[12] = (bf16)f3.x; a.e[13] = (bf16)f3.y;
        a.e[14] = (bf16)f3.z; a.e[15] = (bf16)f3.w;
        af[k] = a.v;
    }
}

// Rebuild one strip's A fragments from the wave-private z slab.
__device__ __forceinline__ void rebuild_afrag(v16bf* af, const bf16* zrows, int lane) {
    const int m = lane & 15, h = lane >> 4;
#pragma unroll
    for (int k = 0; k < KS; ++k) {
        union { v16bf v; v8bf p[2]; } a;
        a.p[0] = *(const v8bf*)(zrows + m * LW + k * 32 + h * 8);
        a.p[1] = *(const v8bf*)(zrows + m * LW + k * 32 + 16 + h * 8);
        af[k] = a.v;
    }
}

// ---------------------------------------------------------------------------
// Kernel 2: fused 3-layer MLP. 8 waves/block, each wave owns 32 rows.
// 6 weight chunks streamed by TDM through 2 LDS buffers; rolling
// s_wait_tensorcnt(1) keeps one DMA in flight behind the active chunk.
// ---------------------------------------------------------------------------
__global__ void __launch_bounds__(256) pcn_mlp(
        const float* __restrict__ x,
        const bf16* __restrict__ wf,
        const float* __restrict__ b1,
        const float* __restrict__ b2,
        const float* __restrict__ b3,
        float* __restrict__ out) {
    __shared__ __align__(64) bf16 wsh0[CHUNK_ELEMS];        // 64 KB
    __shared__ __align__(64) bf16 wsh1[CHUNK_ELEMS];        // 64 KB
    __shared__ __align__(64) bf16 zsh[8][32 * LW];          // 128 KB

    const int tid  = threadIdx.x;
    const int wave = tid >> 5;
    const int lane = tid & 31;
    const int m    = lane & 15;
    bf16* zw = &zsh[wave][0];
    const int row0 = blockIdx.x * 256 + wave * 32;

    // Kick off the first two chunk DMAs; overlap with A-frag build from x.
    if (wave == 0) {
        tdm_load_chunk(wf + 0 * CHUNK_ELEMS, wsh0);
        tdm_load_chunk(wf + 1 * CHUNK_ELEMS, wsh1);
    }

    v16bf afA[KS], afB[KS];
    build_afrag_x(afA, x, row0,      lane);
    build_afrag_x(afB, x, row0 + 16, lane);

    // Per-layer bias values held in registers (one clause, one wait).
    float br[16];
#pragma unroll
    for (int t = 0; t < 16; ++t) br[t] = b1[t * 16 + m];

    if (wave == 0) __builtin_amdgcn_s_wait_tensorcnt(1);   // chunk s0 resident
    __syncthreads();

    // ---- stage 0: layer 0, tiles 0..7 (wsh0) ----
    chunk_relu_lds<0>(wsh0, br, afA, afB, zw, lane);
    __syncthreads();                                       // wsh0 free
    if (wave == 0) {
        tdm_load_chunk(wf + 2 * CHUNK_ELEMS, wsh0);        // stage 2
        __builtin_amdgcn_s_wait_tensorcnt(1);              // chunk s1 resident
    }
    __syncthreads();

    // ---- stage 1: layer 0, tiles 8..15 (wsh1) ----
    chunk_relu_lds<8>(wsh1, br, afA, afB, zw, lane);
    rebuild_afrag(afA, zw,            lane);
    rebuild_afrag(afB, zw + 16 * LW,  lane);
#pragma unroll
    for (int t = 0; t < 16; ++t) br[t] = b2[t * 16 + m];
    __syncthreads();                                       // wsh1 free
    if (wave == 0) {
        tdm_load_chunk(wf + 3 * CHUNK_ELEMS, wsh1);        // stage 3
        __builtin_amdgcn_s_wait_tensorcnt(1);              // chunk s2 resident
    }
    __syncthreads();

    // ---- stage 2: layer 1, tiles 0..7 (wsh0) ----
    chunk_relu_lds<0>(wsh0, br, afA, afB, zw, lane);
    __syncthreads();
    if (wave == 0) {
        tdm_load_chunk(wf + 4 * CHUNK_ELEMS, wsh0);        // stage 4
        __builtin_amdgcn_s_wait_tensorcnt(1);              // chunk s3 resident
    }
    __syncthreads();

    // ---- stage 3: layer 1, tiles 8..15 (wsh1) ----
    chunk_relu_lds<8>(wsh1, br, afA, afB, zw, lane);
    rebuild_afrag(afA, zw,            lane);
    rebuild_afrag(afB, zw + 16 * LW,  lane);
#pragma unroll
    for (int t = 0; t < 16; ++t) br[t] = b3[t * 16 + m];
    __syncthreads();
    if (wave == 0) {
        tdm_load_chunk(wf + 5 * CHUNK_ELEMS, wsh1);        // stage 5
        __builtin_amdgcn_s_wait_tensorcnt(1);              // chunk s4 resident
    }
    __syncthreads();

    // ---- stage 4: layer 2, tiles 0..7 (wsh0) -> out ----
    chunk_out<0>(wsh0, br, afA, afB, out, row0, lane);
    if (wave == 0) __builtin_amdgcn_s_wait_tensorcnt(0);   // chunk s5 resident
    __syncthreads();

    // ---- stage 5: layer 2, tiles 8..15 (wsh1) -> out ----
    chunk_out<8>(wsh1, br, afA, afB, out, row0, lane);
}

// ---------------------------------------------------------------------------
extern "C" void kernel_launch(void* const* d_in, const int* in_sizes, int n_in,
                              void* d_out, int out_size, void* d_ws, size_t ws_size,
                              hipStream_t stream) {
    const float* x  = (const float*)d_in[0];
    const float* p0 = (const float*)d_in[1];
    const float* p1 = (const float*)d_in[2];
    const float* p2 = (const float*)d_in[3];
    const float* p3 = (const float*)d_in[4];
    const float* b1 = (const float*)d_in[5];
    const float* b2 = (const float*)d_in[6];
    const float* b3 = (const float*)d_in[7];

    bf16* wf = (bf16*)d_ws;                          // 3 * 128 KB = 384 KB
    const int welems = 3 * W_ELEMS_PER_LAYER;        // 196608 bf16 elements

    pcn_build_w<<<(welems + 255) / 256, 256, 0, stream>>>(p0, p1, p2, p3, wf);

    pcn_mlp<<<B_ROWS / 256, 256, 0, stream>>>(
        x, wf, b1, b2, b3, (float*)d_out);
}
